// YOLOv3_90804198572445
// MI455X (gfx1250) — matrix operand order repaired and checked by reading, not compiled, not run on hardware
//
#include <hip/hip_runtime.h>

typedef __attribute__((ext_vector_type(16))) _Float16 v16h;
typedef __attribute__((ext_vector_type(8)))  _Float16 v8h;
typedef __attribute__((ext_vector_type(8)))  float    v8f;

#define NIMG 8
#define NCLS 80
#define TOTAL_ANCH 22743
#define FLATSC (TOTAL_ANCH * NCLS)
#define PRE_NMS_N 1000
#define CAND 1024
#define MAX_DET 100
#define SCORE_T 0.05f
#define NMS_T 0.5f
#define IMGSZ 608.0f
#define SCALE_CLAMP_C 4.135166556742356f   /* log(1000/16) */

__constant__ float c_anch[3][3][2] = {
    {{116.f, 90.f}, {156.f, 198.f}, {373.f, 326.f}},
    {{30.f, 61.f},  {62.f, 45.f},   {59.f, 119.f}},
    {{10.f, 13.f},  {16.f, 30.f},   {33.f, 23.f}}};

// ---------------------------------------------------------------------------
// Weight packing: OIHW fp32 (+BN scale fold) -> fragment-major f16:
//   wBp[kt][ntile][lane][e]  (kt = 32-deep K tile, k = (kh*K+kw)*Cin+ic)
//   lane 0-15: K = kt*32 + {0..7, 16..23}[e]; lane 16-31: K = kt*32 + {8..15, 24..31}[e]
// ---------------------------------------------------------------------------
__global__ void pack_w(const float* __restrict__ w, const float* __restrict__ s,
                       _Float16* __restrict__ wBp, int OC, int Cin, int K, int OCp)
{
    long long idx = (long long)blockIdx.x * 256 + threadIdx.x;
    long long Kt = (long long)K * K * Cin;             // multiple of 32
    if (idx >= Kt * OCp) return;
    int e    = (int)(idx & 15);
    int lane = (int)((idx >> 4) & 31);
    long long rest = idx >> 9;
    int NT = OCp >> 4;
    int nt = (int)(rest % NT);
    long long kt = rest / NT;
    long long k = kt * 32 + (e >> 3) * 16 + (lane >> 4) * 8 + (e & 7);
    int oc = nt * 16 + (lane & 15);
    float v = 0.f;
    if (oc < OC) {
        int ic = (int)(k % Cin);
        int t  = (int)(k / Cin);
        int kw = t % K, kh = t / K;
        v = w[(((size_t)oc * Cin + ic) * K + kh) * K + kw];
        if (s) v *= s[oc];
    }
    wBp[idx] = (_Float16)v;
}

// ---------------------------------------------------------------------------
// Layout conversions
// ---------------------------------------------------------------------------
__global__ void nchw_to_nhwc(const float* __restrict__ src, _Float16* __restrict__ dst,
                             int Nn, int C, int H, int W, int Ctot, int cOff)
{
    long long idx = (long long)blockIdx.x * 256 + threadIdx.x;
    long long total = (long long)Nn * C * H * W;
    if (idx >= total) return;
    int x = (int)(idx % W);
    int y = (int)((idx / W) % H);
    int c = (int)((idx / ((long long)W * H)) % C);
    int n = (int)(idx / ((long long)W * H * C));
    dst[(((size_t)n * H + y) * W + x) * Ctot + cOff + c] = (_Float16)src[idx];
}

__global__ void upsample2x(const _Float16* __restrict__ src, _Float16* __restrict__ dst,
                           int Nn, int H, int W, int C, int Ctot)
{
    long long idx = (long long)blockIdx.x * 256 + threadIdx.x;
    int H2 = 2 * H, W2 = 2 * W;
    long long total = (long long)Nn * H2 * W2 * C;
    if (idx >= total) return;
    int c = (int)(idx % C);
    int x = (int)((idx / C) % W2);
    int y = (int)((idx / ((long long)C * W2)) % H2);
    int n = (int)(idx / ((long long)C * W2 * H2));
    dst[(((size_t)n * H2 + y) * W2 + x) * Ctot + c] =
        src[(((size_t)n * H + (y >> 1)) * W + (x >> 1)) * C + c];
}

// ---------------------------------------------------------------------------
// Implicit-GEMM conv (stride 1, SAME) via v_wmma_f32_16x16x32_f16.
// 4 waves / block; each wave: 32 output pixels x 64 output channels
// (2 A fragments x 4 B fragments = 8 WMMAs per 32-deep k-step).
// All fragment loads are 16-byte vector loads; no div/mod in the k-loop.
// ---------------------------------------------------------------------------
__global__ __launch_bounds__(128)
void conv_wmma(const _Float16* __restrict__ src, const _Float16* __restrict__ wBp,
               const float* __restrict__ bias,
               _Float16* __restrict__ dst16, float* __restrict__ dst32,
               int Nn, int H, int W, int Cin, int OC, int OCp, int K, int leaky)
{
    const int lane = threadIdx.x & 31;
    const int wave = threadIdx.x >> 5;
    const int half = lane >> 4;
    const int l16  = lane & 15;
    const int M    = Nn * H * W;
    const int m0   = blockIdx.x * 128 + wave * 32;
    const int n0   = blockIdx.y * 64;
    const int pad  = K / 2;
    const int NT   = OCp >> 4;          // total 16-col tiles in packed weights
    const int ntb  = n0 >> 4;           // first col tile for this block

    int imgA[2], yyA[2], xxA[2];
    bool rowOK[2];
#pragma unroll
    for (int mt = 0; mt < 2; ++mt) {
        int arow = m0 + mt * 16 + l16;
        rowOK[mt] = arow < M;
        int img = 0, yy = 0, xx = 0;
        if (rowOK[mt]) {
            img = arow / (H * W);
            int r = arow % (H * W);
            yy = r / W; xx = r % W;
        }
        imgA[mt] = img; yyA[mt] = yy; xxA[mt] = xx;
    }

    v8f acc[2][4] = {};
    int kt = 0;
    for (int kh = 0; kh < K; ++kh) {
        for (int kw = 0; kw < K; ++kw) {
            const _Float16* srow[2];
            bool ok[2];
#pragma unroll
            for (int mt = 0; mt < 2; ++mt) {
                int y = yyA[mt] + kh - pad, x = xxA[mt] + kw - pad;
                ok[mt] = rowOK[mt] && (unsigned)y < (unsigned)H && (unsigned)x < (unsigned)W;
                srow[mt] = src + (((size_t)imgA[mt] * H + y) * W + x) * Cin + half * 8;
            }
            for (int ic0 = 0; ic0 < Cin; ic0 += 32, ++kt) {
                v16h a[2];
#pragma unroll
                for (int mt = 0; mt < 2; ++mt) {
                    v8h lo = {}, hi = {};
                    if (ok[mt]) {
                        lo = *(const v8h*)(srow[mt] + ic0);       // K = kb+0..7
                        hi = *(const v8h*)(srow[mt] + ic0 + 16);  // K = kb+16..23
                    }
#pragma unroll
                    for (int e = 0; e < 8; ++e) { a[mt][e] = lo[e]; a[mt][8 + e] = hi[e]; }
                }
                const _Float16* bp = wBp + (((size_t)kt * NT + ntb) * 32 + lane) * 16;
#pragma unroll
                for (int nt = 0; nt < 4; ++nt) {
                    v8h blo = *(const v8h*)(bp);
                    v8h bhi = *(const v8h*)(bp + 8);
                    bp += 512;                                    // next 16-col tile
                    v16h b;
#pragma unroll
                    for (int e = 0; e < 8; ++e) { b[e] = blo[e]; b[8 + e] = bhi[e]; }
                    acc[0][nt] = __builtin_amdgcn_wmma_f32_16x16x32_f16(
                        false, a[0], false, b, (short)0, acc[0][nt], false, false);
                    acc[1][nt] = __builtin_amdgcn_wmma_f32_16x16x32_f16(
                        false, a[1], false, b, (short)0, acc[1][nt], false, false);
                }
            }
        }
    }

    // ---- epilogue: C layout: lane -> N=l16, VGPR v -> M = base + v + 8*half
#pragma unroll
    for (int mt = 0; mt < 2; ++mt) {
#pragma unroll
        for (int nt = 0; nt < 4; ++nt) {
            const int col = n0 + nt * 16 + l16;
            if (col >= OC) continue;
            const float bi = bias ? bias[col] : 0.f;
#pragma unroll
            for (int v = 0; v < 8; ++v) {
                const int m = m0 + mt * 16 + v + half * 8;
                if (m >= M) continue;
                float val = acc[mt][nt][v] + bi;
                if (leaky) val = val > 0.f ? val : 0.1f * val;
                if (dst16) dst16[(size_t)m * OC + col] = (_Float16)val;
                if (dst32) dst32[(size_t)m * OC + col] = val;
            }
        }
    }
}

// ---------------------------------------------------------------------------
// Decode anchors -> boxes (clipped) + obj*cls scores
// ---------------------------------------------------------------------------
__global__ void decode_kernel(const float* __restrict__ p0, const float* __restrict__ p1,
                              const float* __restrict__ p2,
                              float* __restrict__ boxes, float* __restrict__ scores)
{
    long long idx = (long long)blockIdx.x * 256 + threadIdx.x;
    if (idx >= (long long)NIMG * TOTAL_ANCH) return;
    int img = (int)(idx / TOTAL_ANCH);
    int a0  = (int)(idx % TOTAL_ANCH);
    int a = a0, lvl, G;
    float stride;
    const float* p;
    if (a < 1083)      { p = p0; G = 19; stride = 32.f; lvl = 0; }
    else if (a < 5415) { p = p1; G = 38; stride = 16.f; lvl = 1; a -= 1083; }
    else               { p = p2; G = 76; stride = 8.f;  lvl = 2; a -= 5415; }
    int ai = a % 3, cell = a / 3;
    int x = cell % G, y = cell / G;
    const float* pp = p + ((((size_t)img * G + y) * G + x) * 255) + (size_t)ai * 85;
    float aw = c_anch[lvl][ai][0], ah = c_anch[lvl][ai][1];
    float cx = (x + 0.5f) * stride, cy = (y + 0.5f) * stride;
    float dx = pp[0], dy = pp[1];
    float dw = fminf(pp[2], SCALE_CLAMP_C), dh = fminf(pp[3], SCALE_CLAMP_C);
    float pcx = dx * aw + cx, pcy = dy * ah + cy;
    float pw = __expf(dw) * aw, ph = __expf(dh) * ah;
    size_t bb = ((size_t)img * TOTAL_ANCH + a0) * 4;
    boxes[bb + 0] = fminf(fmaxf(pcx - 0.5f * pw, 0.f), IMGSZ);
    boxes[bb + 1] = fminf(fmaxf(pcy - 0.5f * ph, 0.f), IMGSZ);
    boxes[bb + 2] = fminf(fmaxf(pcx + 0.5f * pw, 0.f), IMGSZ);
    boxes[bb + 3] = fminf(fmaxf(pcy + 0.5f * ph, 0.f), IMGSZ);
    float obj = 1.f / (1.f + __expf(-pp[4]));
    size_t sb = ((size_t)img * TOTAL_ANCH + a0) * NCLS;
    for (int c = 0; c < NCLS; ++c)
        scores[sb + c] = obj * (1.f / (1.f + __expf(-pp[5 + c])));
}

// ---------------------------------------------------------------------------
// Top-K selection machinery
// ---------------------------------------------------------------------------
__global__ void zero_u32(unsigned* __restrict__ p, long long n)
{
    long long i = (long long)blockIdx.x * 256 + threadIdx.x;
    if (i < n) p[i] = 0u;
}

__global__ void hist_kernel(const float* __restrict__ scores, unsigned* __restrict__ hist)
{
    long long idx = (long long)blockIdx.x * 256 + threadIdx.x;
    if (idx >= (long long)NIMG * FLATSC) return;
    int img = (int)(idx / FLATSC);
    float s = scores[idx];
    if (s > SCORE_T) {
        unsigned b = __float_as_uint(s) >> 22;   // non-negative floats: bits monotonic
        atomicAdd(&hist[img * 1024 + b], 1u);
    }
}

__global__ void thresh_kernel(const unsigned* __restrict__ hist, unsigned* __restrict__ thr)
{
    if (threadIdx.x != 0) return;
    int img = blockIdx.x;
    unsigned cum = 0;
    int b;
    for (b = 1023; b >= 0; --b) {
        cum += hist[img * 1024 + b];
        if (cum >= PRE_NMS_N) break;
    }
    if (b < 0) b = 0;
    thr[img] = (unsigned)b << 22;
}

__global__ void compact_kernel(const float* __restrict__ scores, const unsigned* __restrict__ thr,
                               unsigned* __restrict__ cnt, float* __restrict__ candS,
                               unsigned* __restrict__ candP)
{
    long long idx = (long long)blockIdx.x * 256 + threadIdx.x;
    if (idx >= (long long)NIMG * FLATSC) return;
    int img = (int)(idx / FLATSC);
    int f   = (int)(idx % FLATSC);
    float s = scores[idx];
    if (s > SCORE_T && __float_as_uint(s) >= thr[img]) {
        unsigned pos = atomicAdd(&cnt[img], 1u);
        if (pos < PRE_NMS_N) {
            candS[img * CAND + pos] = s;
            candP[img * CAND + pos] = (unsigned)f;   // anchor*80 + class
        }
    }
}

// ---------------------------------------------------------------------------
// Per-image: bitonic sort (desc) + class-offset greedy NMS + emit top-100
// ---------------------------------------------------------------------------
__global__ __launch_bounds__(512)
void nms_kernel(const float* __restrict__ candS, const unsigned* __restrict__ candP,
                const float* __restrict__ boxes, float* __restrict__ dets,
                int* __restrict__ classes)
{
    const int img = blockIdx.x;
    __shared__ float    key[CAND];
    __shared__ unsigned pay[CAND];
    __shared__ float    bx[PRE_NMS_N][4];
    __shared__ unsigned char keep[PRE_NMS_N];

    for (int i = threadIdx.x; i < CAND; i += 512) {
        key[i] = candS[img * CAND + i];
        pay[i] = candP[img * CAND + i];
    }
    __syncthreads();

    for (int k = 2; k <= CAND; k <<= 1) {
        for (int j = k >> 1; j > 0; j >>= 1) {
            for (int i = threadIdx.x; i < CAND; i += 512) {
                int ixj = i ^ j;
                if (ixj > i) {
                    bool up = ((i & k) == 0);           // descending order
                    float a = key[i], b = key[ixj];
                    if (up ? (a < b) : (a > b)) {
                        key[i] = b; key[ixj] = a;
                        unsigned t = pay[i]; pay[i] = pay[ixj]; pay[ixj] = t;
                    }
                }
            }
            __syncthreads();
        }
    }

    for (int i = threadIdx.x; i < PRE_NMS_N; i += 512) {
        unsigned f = pay[i];
        int anc = (int)(f / NCLS), cls = (int)(f % NCLS);
        const float* b = boxes + ((size_t)img * TOTAL_ANCH + anc) * 4;
        float off = cls * (IMGSZ + 1.0f);
        bx[i][0] = b[0] + off; bx[i][1] = b[1] + off;
        bx[i][2] = b[2] + off; bx[i][3] = b[3] + off;
        keep[i] = (key[i] > 0.f) ? 1 : 0;
    }
    __syncthreads();

    for (int i = 0; i < PRE_NMS_N; ++i) {
        if (keep[i]) {
            float x1 = bx[i][0], y1 = bx[i][1], x2 = bx[i][2], y2 = bx[i][3];
            float ai = (x2 - x1) * (y2 - y1);
            for (int jj = threadIdx.x; jj < PRE_NMS_N; jj += 512) {
                if (jj > i && keep[jj]) {
                    float xx1 = fmaxf(x1, bx[jj][0]), yy1 = fmaxf(y1, bx[jj][1]);
                    float xx2 = fminf(x2, bx[jj][2]), yy2 = fminf(y2, bx[jj][3]);
                    float w = fmaxf(xx2 - xx1, 0.f), h = fmaxf(yy2 - yy1, 0.f);
                    float inter = w * h;
                    float aj = (bx[jj][2] - bx[jj][0]) * (bx[jj][3] - bx[jj][1]);
                    if (inter / (ai + aj - inter + 1e-9f) > NMS_T) keep[jj] = 0;
                }
            }
        }
        __syncthreads();
    }

    if (threadIdx.x == 0) {
        float* det = dets + (size_t)img * MAX_DET * 5;
        int* co = classes + img * MAX_DET;
        int n = 0;
        for (int i = 0; i < PRE_NMS_N && n < MAX_DET; ++i) {
            if (keep[i] && key[i] > 0.f) {
                unsigned f = pay[i];
                int anc = (int)(f / NCLS), cls = (int)(f % NCLS);
                const float* b = boxes + ((size_t)img * TOTAL_ANCH + anc) * 4;
                det[n * 5 + 0] = b[0]; det[n * 5 + 1] = b[1];
                det[n * 5 + 2] = b[2]; det[n * 5 + 3] = b[3];
                det[n * 5 + 4] = key[i];
                co[n] = cls; ++n;
            }
        }
        for (; n < MAX_DET; ++n) {
            det[n * 5 + 0] = det[n * 5 + 1] = det[n * 5 + 2] = det[n * 5 + 3] = det[n * 5 + 4] = 0.f;
            co[n] = 0;
        }
    }
}

// ---------------------------------------------------------------------------
// Host orchestration
// ---------------------------------------------------------------------------
extern "C" void kernel_launch(void* const* d_in, const int* in_sizes, int n_in,
                              void* d_out, int out_size, void* d_ws, size_t ws_size,
                              hipStream_t stream)
{
    (void)in_sizes; (void)n_in; (void)out_size; (void)ws_size;

    // ---- input unpack (jax dict-flatten order: f3,f4,f5, params{blocks,pred,up}, CBL keys b,s,w)
    const float* f3 = (const float*)d_in[0];
    const float* f4 = (const float*)d_in[1];
    const float* f5 = (const float*)d_in[2];
    struct CBLW { const float *b, *s, *w; };
    CBLW blk[3][5];
    int p = 3;
    for (int i = 0; i < 3; ++i)
        for (int j = 0; j < 5; ++j) {
            blk[i][j].b = (const float*)d_in[p];
            blk[i][j].s = (const float*)d_in[p + 1];
            blk[i][j].w = (const float*)d_in[p + 2];
            p += 3;
        }
    const float* pred_b1[3]; CBLW pred_c0[3]; const float* pred_w1[3];
    for (int i = 0; i < 3; ++i) {
        pred_b1[i] = (const float*)d_in[p];
        pred_c0[i].b = (const float*)d_in[p + 1];
        pred_c0[i].s = (const float*)d_in[p + 2];
        pred_c0[i].w = (const float*)d_in[p + 3];
        pred_w1[i]  = (const float*)d_in[p + 4];
        p += 5;
    }
    CBLW up[2];
    for (int i = 0; i < 2; ++i) {
        up[i].b = (const float*)d_in[p];
        up[i].s = (const float*)d_in[p + 1];
        up[i].w = (const float*)d_in[p + 2];
        p += 3;
    }

    // ---- workspace bump allocator
    char* base = (char*)d_ws;
    size_t off = 0;
    auto alloc = [&](size_t bytes) -> void* {
        void* r = base + off;
        off += (bytes + 255) & ~(size_t)255;
        return r;
    };
    _Float16* wbuf  = (_Float16*)alloc((size_t)4608 * 1024 * 2);  // max packed weight
    _Float16* cat0  = (_Float16*)alloc((size_t)NIMG * 19 * 19 * 1024 * 2);
    _Float16* cat1  = (_Float16*)alloc((size_t)NIMG * 38 * 38 * 768 * 2);
    _Float16* cat2  = (_Float16*)alloc((size_t)NIMG * 76 * 76 * 384 * 2);
    _Float16* tmpA  = (_Float16*)alloc((size_t)NIMG * 76 * 76 * 256 * 2);
    _Float16* tmpB  = (_Float16*)alloc((size_t)NIMG * 76 * 76 * 256 * 2);
    _Float16* upb   = (_Float16*)alloc((size_t)NIMG * 38 * 38 * 128 * 2);
    float* pred0 = (float*)alloc((size_t)NIMG * 19 * 19 * 255 * 4);
    float* pred1 = (float*)alloc((size_t)NIMG * 38 * 38 * 255 * 4);
    float* pred2 = (float*)alloc((size_t)NIMG * 76 * 76 * 255 * 4);
    float* preds[3] = {pred0, pred1, pred2};
    float* boxes  = (float*)alloc((size_t)NIMG * TOTAL_ANCH * 4 * 4);
    float* scores = (float*)alloc((size_t)NIMG * (size_t)FLATSC * 4);
    unsigned* hist  = (unsigned*)alloc((size_t)NIMG * 1024 * 4);
    unsigned* thr   = (unsigned*)alloc((size_t)NIMG * 4);
    unsigned* cnt   = (unsigned*)alloc((size_t)NIMG * 4);
    float*    candS = (float*)alloc((size_t)NIMG * CAND * 4);
    unsigned* candP = (unsigned*)alloc((size_t)NIMG * CAND * 4);

    auto conv = [&](const _Float16* src, const float* wraw, const float* scale,
                    const float* bias, _Float16* dst16, float* dst32,
                    int H, int Cin, int OC, int K, int leaky) {
        int OCp = (OC + 63) & ~63;
        long long tw = (long long)K * K * Cin * OCp;
        pack_w<<<dim3((unsigned)((tw + 255) / 256)), dim3(256), 0, stream>>>(
            wraw, scale, wbuf, OC, Cin, K, OCp);
        int M = NIMG * H * H;
        dim3 g((M + 127) / 128, OCp / 64);
        conv_wmma<<<g, dim3(128), 0, stream>>>(src, wbuf, bias, dst16, dst32,
                                               NIMG, H, H, Cin, OC, OCp, K, leaky);
    };

    const int Gs[3]  = {19, 38, 76};
    const int OCs[3] = {512, 256, 128};
    const int ICs[3] = {1024, 768, 384};
    _Float16* cats[3] = {cat0, cat1, cat2};

    {   // lvl0 input: f5 -> NHWC f16
        long long tot = (long long)NIMG * 1024 * 19 * 19;
        nchw_to_nhwc<<<dim3((unsigned)((tot + 255) / 256)), dim3(256), 0, stream>>>(
            f5, cat0, NIMG, 1024, 19, 19, 1024, 0);
    }

    for (int i = 0; i < 3; ++i) {
        int G = Gs[i], oc = OCs[i], ic = ICs[i];
        conv(cats[i], blk[i][0].w, blk[i][0].s, blk[i][0].b, tmpA, nullptr, G, ic,     oc,     1, 1);
        conv(tmpA,    blk[i][1].w, blk[i][1].s, blk[i][1].b, tmpB, nullptr, G, oc,     2 * oc, 3, 1);
        conv(tmpB,    blk[i][2].w, blk[i][2].s, blk[i][2].b, tmpA, nullptr, G, 2 * oc, oc,     1, 1);
        conv(tmpA,    blk[i][3].w, blk[i][3].s, blk[i][3].b, tmpB, nullptr, G, oc,     2 * oc, 3, 1);
        conv(tmpB,    blk[i][4].w, blk[i][4].s, blk[i][4].b, tmpA, nullptr, G, 2 * oc, oc,     1, 1);
        // prediction head
        conv(tmpA, pred_c0[i].w, pred_c0[i].s, pred_c0[i].b, tmpB, nullptr, G, oc, 2 * oc, 3, 1);
        conv(tmpB, pred_w1[i], nullptr, pred_b1[i], nullptr, preds[i], G, 2 * oc, 255, 1, 0);
        if (i < 2) {
            conv(tmpA, up[i].w, up[i].s, up[i].b, upb, nullptr, G, oc, oc / 2, 1, 1);
            long long tu = (long long)NIMG * 2 * G * 2 * G * (oc / 2);
            upsample2x<<<dim3((unsigned)((tu + 255) / 256)), dim3(256), 0, stream>>>(
                upb, cats[i + 1], NIMG, G, G, oc / 2, ICs[i + 1]);
            const float* fsrc = (i == 0) ? f4 : f3;
            int fc = (i == 0) ? 512 : 256;
            long long tc = (long long)NIMG * fc * 2 * G * 2 * G;
            nchw_to_nhwc<<<dim3((unsigned)((tc + 255) / 256)), dim3(256), 0, stream>>>(
                fsrc, cats[i + 1], NIMG, fc, 2 * G, 2 * G, ICs[i + 1], oc / 2);
        }
    }

    // ---- decode
    {
        long long tot = (long long)NIMG * TOTAL_ANCH;
        decode_kernel<<<dim3((unsigned)((tot + 255) / 256)), dim3(256), 0, stream>>>(
            pred0, pred1, pred2, boxes, scores);
    }
    // ---- zero NMS scratch
    zero_u32<<<dim3((NIMG * 1024 + 255) / 256), dim3(256), 0, stream>>>(hist, (long long)NIMG * 1024);
    zero_u32<<<dim3(1), dim3(256), 0, stream>>>(cnt, NIMG);
    zero_u32<<<dim3((NIMG * CAND + 255) / 256), dim3(256), 0, stream>>>((unsigned*)candS, (long long)NIMG * CAND);
    zero_u32<<<dim3((NIMG * CAND + 255) / 256), dim3(256), 0, stream>>>(candP, (long long)NIMG * CAND);
    // ---- top-k: histogram -> threshold -> compact
    {
        long long tot = (long long)NIMG * FLATSC;
        hist_kernel<<<dim3((unsigned)((tot + 255) / 256)), dim3(256), 0, stream>>>(scores, hist);
        thresh_kernel<<<dim3(NIMG), dim3(32), 0, stream>>>(hist, thr);
        compact_kernel<<<dim3((unsigned)((tot + 255) / 256)), dim3(256), 0, stream>>>(
            scores, thr, cnt, candS, candP);
    }
    // ---- sort + NMS + emit
    float* dets = (float*)d_out;
    int* classes = (int*)d_out + NIMG * MAX_DET * 5;
    nms_kernel<<<dim3(NIMG), dim3(512), 0, stream>>>(candS, candP, boxes, dets, classes);
}
